// MultiHeadSelfAttention_1486058684810
// MI455X (gfx1250) — matrix operand order repaired and checked by reading, not compiled, not run on hardware
//
#include <hip/hip_runtime.h>
#include <hip/hip_bf16.h>

// ---------------------------------------------------------------------------
// MHA (buggy-reshape variant) for MI455X / gfx1250.
//  - all matmuls via v_wmma_f32_16x16x32_bf16 (bf16 in, f32 accumulate)
//  - attention K-chunk staged by the Tensor Data Mover (tensor_load_to_lds,
//    TENSORcnt) with LDS row padding expressed in the descriptor
//  - attention Q tile staged with GLOBAL_LOAD_ASYNC_TO_LDS_B128 (ASYNCcnt)
//  - softmax row reductions via v_permlane16_b32 (no LDS round-trips)
//  - softmax via raw v_exp_f32 (__builtin_amdgcn_exp2f), log2e folded into Q
// Roofline: ~60 GFLOP, ~50 MB working set (fits in 192 MB L2) -> matrix-core
// bound, so everything routes through the bf16 WMMA path.
// ---------------------------------------------------------------------------

typedef __attribute__((ext_vector_type(16))) __bf16        v16bf;
typedef __attribute__((ext_vector_type(8)))  float         v8f;
typedef __attribute__((ext_vector_type(4)))  unsigned int  v4u;
typedef __attribute__((ext_vector_type(8)))  int           v8i;
typedef __attribute__((ext_vector_type(4)))  int           v4i;

#define AS_GLOBAL __attribute__((address_space(1)))
#define AS_LDS    __attribute__((address_space(3)))

static constexpr int BB = 2, S = 2048, E = 1024, H = 16, Dh = 64;
// softmax scale 1/sqrt(64)=0.125 folded into Q, plus log2(e) so the
// attention kernel can use base-2 exponentials (v_exp_f32) directly.
static constexpr float QSCALE = 0.125f * 1.4426950408889634f;

#if __has_builtin(__builtin_amdgcn_exp2f)
#define fast_exp2 __builtin_amdgcn_exp2f
#else
#define fast_exp2 exp2f
#endif

#if __has_builtin(__builtin_amdgcn_global_load_async_to_lds_b128)
#define HAVE_ASYNC_LDS 1
#else
#define HAVE_ASYNC_LDS 0
#endif

#if __has_builtin(__builtin_amdgcn_tensor_load_to_lds)
#define HAVE_TDM 1
#else
#define HAVE_TDM 0
#endif

#if __has_builtin(__builtin_amdgcn_permlane16)
#define HAVE_PERMLANE16 1
#else
#define HAVE_PERMLANE16 0
#endif

__device__ __forceinline__ void wait_asynccnt0() {
#if __has_builtin(__builtin_amdgcn_s_wait_asynccnt)
  __builtin_amdgcn_s_wait_asynccnt(0);
#else
  asm volatile("s_wait_asynccnt 0" ::: "memory");
#endif
}

__device__ __forceinline__ void wait_tensorcnt0() {
#if __has_builtin(__builtin_amdgcn_s_wait_tensorcnt)
  __builtin_amdgcn_s_wait_tensorcnt(0);
#else
  asm volatile("s_wait_tensorcnt 0" ::: "memory");
#endif
}

// xor-shuffle within 16-lane groups (the v_permlane16 "row" == our reduction
// group). Selector nibble for lane j is (j^M)&15, same pattern in both rows.
template <int M>
__device__ __forceinline__ float lane16_xor(float v) {
#if HAVE_PERMLANE16
  unsigned lo, hi;
  if constexpr (M == 1)      { lo = 0x67452301u; hi = 0xEFCDAB89u; }
  else if constexpr (M == 2) { lo = 0x54761032u; hi = 0xDCFE98BAu; }
  else if constexpr (M == 4) { lo = 0x32107654u; hi = 0xBA98FEDCu; }
  else                       { lo = 0xFEDCBA98u; hi = 0x76543210u; } // M==8
  unsigned x = __builtin_amdgcn_permlane16(
      __float_as_uint(v), __float_as_uint(v), lo, hi, false, false);
  return __uint_as_float(x);
#else
  return __shfl_xor(v, M, 32);
#endif
}

__device__ __forceinline__ v8f zero8() {
  v8f z = {0.f, 0.f, 0.f, 0.f, 0.f, 0.f, 0.f, 0.f};
  return z;
}

// ---------------------------------------------------------------------------
// GEMM:  out[M,N] = (A[M,K] @ W[K,N] + bias[N]) * scale
// Block tile 128x64, BK=32. 8 waves; each wave computes a 32x32 subtile
// (2x2 WMMA accumulators). Grid: (N/64, M/128).
// ---------------------------------------------------------------------------
template <bool A_BF16, bool OUT_BF16>
__global__ __launch_bounds__(256)
void gemm_bias_wmma(const void* __restrict__ Ap, const float* __restrict__ W,
                    const float* __restrict__ bias, void* __restrict__ outp,
                    int M, int N, int K, float scale) {
  constexpr int BM = 128, BN = 64, BK = 32, PAD = 2;
  __shared__ __align__(16) __bf16 As[BM][BK + PAD];  // [m][k]
  __shared__ __align__(16) __bf16 Bs[BN][BK + PAD];  // [n][k] (transposed)

  const int tid  = threadIdx.x;
  const int lane = tid & 31;
  const int wv   = tid >> 5;            // 0..7
  const int wr   = wv & 3;              // 4 wave-rows of 32
  const int wc   = wv >> 2;             // 2 wave-cols of 32
  const int ln   = lane & 15;
  const int hi   = lane >> 4;           // half-wave select
  const int m0   = blockIdx.y * BM;
  const int n0   = blockIdx.x * BN;

  const float*  Af = (const float*)Ap;
  const __bf16* Ab = (const __bf16*)Ap;

  v8f acc[2][2];
#pragma unroll
  for (int mi = 0; mi < 2; ++mi)
#pragma unroll
    for (int ni = 0; ni < 2; ++ni) acc[mi][ni] = zero8();

  for (int k0 = 0; k0 < K; k0 += BK) {
    // cooperative load of A tile (convert fp32 -> bf16 on the fly)
    for (int i = tid; i < BM * BK; i += 256) {
      int r = i >> 5, c = i & 31;
      float v;
      if constexpr (A_BF16)
        v = (float)Ab[(size_t)(m0 + r) * K + k0 + c];
      else
        v = Af[(size_t)(m0 + r) * K + k0 + c];
      As[r][c] = (__bf16)v;
    }
    // cooperative load of W tile, transposed into LDS as [n][k]
    for (int i = tid; i < BK * BN; i += 256) {
      int k = i >> 6, n = i & 63;
      Bs[n][k] = (__bf16)W[(size_t)(k0 + k) * N + n0 + n];
    }
    __syncthreads();

    // A fragment: lane<16 holds M=ln, K {0..15}; lane>=16: K {16..31}
    v16bf afr[2], bfr[2];
    const int khA = hi * 16;
    const int khB = hi * 16;
#pragma unroll
    for (int mi = 0; mi < 2; ++mi) {
      const int row = wr * 32 + mi * 16 + ln;
#pragma unroll
      for (int i = 0; i < 16; ++i)
        afr[mi][i] = As[row][khA + i];
    }
#pragma unroll
    for (int ni = 0; ni < 2; ++ni) {
      const int col = wc * 32 + ni * 16 + ln;
#pragma unroll
      for (int i = 0; i < 16; ++i) bfr[ni][i] = Bs[col][khB + i];
    }
#pragma unroll
    for (int mi = 0; mi < 2; ++mi)
#pragma unroll
      for (int ni = 0; ni < 2; ++ni)
        acc[mi][ni] = __builtin_amdgcn_wmma_f32_16x16x32_bf16(
            false, afr[mi], false, bfr[ni], (short)0, acc[mi][ni], false, false);
    __syncthreads();
  }

  // epilogue: D layout -> VGPR r holds row (r + 8*hi), col = ln
  float*  outf = (float*)outp;
  __bf16* outb = (__bf16*)outp;
#pragma unroll
  for (int mi = 0; mi < 2; ++mi)
#pragma unroll
    for (int ni = 0; ni < 2; ++ni) {
      const int col = n0 + wc * 32 + ni * 16 + ln;
      const float bv = bias[col];
#pragma unroll
      for (int r = 0; r < 8; ++r) {
        const int row = m0 + wr * 32 + mi * 16 + r + hi * 8;
        const float v = (acc[mi][ni][r] + bv) * scale;
        if constexpr (OUT_BF16)
          outb[(size_t)row * N + col] = (__bf16)v;
        else
          outf[(size_t)row * N + col] = v;
      }
    }
}

// ---------------------------------------------------------------------------
// Flash attention per (b,h). Q pre-scaled by 0.125*log2(e) -> exp2 softmax.
// Block: one (b,h), 128 Q rows; wave w owns rows w*16..+15. KV chunks of 64.
// K chunk staged by TDM (descriptor carries the LDS row padding), Q tile by
// async-to-LDS copies. Grid: (S/128, B*H).
// ---------------------------------------------------------------------------
__global__ __launch_bounds__(256)
void flash_attn_wmma(const __bf16* __restrict__ Qb, const __bf16* __restrict__ Kb,
                     const __bf16* __restrict__ Vb, __bf16* __restrict__ Xh) {
  constexpr int BQ = 128, BKV = 64, PAD = 8;
  __shared__ __align__(16) __bf16 Qs[BQ][Dh + PAD];   // [q][d]
  __shared__ __align__(16) __bf16 Ks[BKV][Dh + PAD];  // [kv][d]
  __shared__ __align__(16) __bf16 Vt[Dh][BKV + PAD];  // [d][kv] (V transposed)
  __shared__ __align__(16) __bf16 Ps[BQ][BKV + PAD];  // [q][kv] (P bounce)

  const int tid  = threadIdx.x;
  const int lane = tid & 31;
  const int wv   = tid >> 5;
  const int ln   = lane & 15;
  const int hi   = lane >> 4;
  const int bh   = blockIdx.y;           // b*H + h
  const int q0   = blockIdx.x * BQ;

  const size_t  base = (size_t)bh * S * Dh;   // contiguous [2048 x 64] per head
  const __bf16* Qh = Qb + base;
  const __bf16* Kh = Kb + base;
  const __bf16* Vh = Vb + base;

  // ---- load Q tile -------------------------------------------------------
#if HAVE_ASYNC_LDS
  // 128 rows x 128B, 16B per lane-op; LDS row stride 144B (pad preserved).
  for (int i = tid; i < BQ * 8; i += 256) {
    int r = i >> 3, cp = i & 7;
    __builtin_amdgcn_global_load_async_to_lds_b128(
        (AS_GLOBAL v4i*)(void*)(Qh + (size_t)(q0 + r) * Dh + cp * 8),
        (AS_LDS v4i*)&Qs[r][cp * 8], 0, 0);
  }
  wait_asynccnt0();
#else
  for (int i = tid; i < BQ * (Dh / 2); i += 256) {
    int r = i >> 5, cp = i & 31;
    *(unsigned int*)&Qs[r][cp * 2] =
        ((const unsigned int*)Qh)[(size_t)(q0 + r) * (Dh / 2) + cp];
  }
#endif
  __syncthreads();

  // persistent Q A-fragments (two K-steps of the d=64 dimension)
  v16bf qa[2];
  {
    const int row = wv * 16 + ln;
#pragma unroll
    for (int ks = 0; ks < 2; ++ks)
#pragma unroll
      for (int i = 0; i < 16; ++i)
        qa[ks][i] = Qs[row][ks * 32 + hi * 16 + i];
  }

  v8f o[4];
#pragma unroll
  for (int j = 0; j < 4; ++j) o[j] = zero8();
  float mrow[8], lrow[8];
#pragma unroll
  for (int r = 0; r < 8; ++r) { mrow[r] = -1e30f; lrow[r] = 0.f; }

  for (int c0 = 0; c0 < S; c0 += BKV) {
    // ---- stage K chunk [64 x 64] bf16 into Ks ---------------------------
#if HAVE_TDM
    if (wv == 0) {
      // Tensor DMA descriptor: 2D tensor [2048 x 64] of 2B elements,
      // tile 64x64 at row c0; LDS padding 4 DWORDs every 32 DWORDs
      // (-> row stride 144B == Ks[64][72]).
      const unsigned long long ga =
          (unsigned long long)(uintptr_t)(Kh + (size_t)c0 * Dh);
      const unsigned lds_off = (unsigned)(uintptr_t)(AS_LDS char*)&Ks[0][0];
      v4u g0;
      g0[0] = 1u;                                            // count=1 (valid)
      g0[1] = lds_off;                                       // lds_addr
      g0[2] = (unsigned)ga;                                  // global_addr lo
      g0[3] = (unsigned)((ga >> 32) & 0x01FFFFFFu) | (2u << 30);  // hi | type=2
      v8i g1;
      g1[0] = (int)((1u << 16)        // data_size = 2B
                    | (1u << 20)      // pad_enable
                    | (4u << 22)      // pad_interval: 32 DWORDs (128B)
                    | (3u << 25));    // pad_amount: 4 DWORDs (16B)
      g1[1] = (int)(64u << 16);       // tensor_dim0 = 64 (lo16 -> [31:16])
      g1[2] = (int)(2048u << 16);     // dim0 hi16=0 | tensor_dim1=2048 lo16
      g1[3] = (int)(64u << 16);       // dim1 hi16=0 | tile_dim0 = 64
      g1[4] = 64;                     // tile_dim1 = 64 | tile_dim2 = 0
      g1[5] = 64;                     // tensor_dim0_stride = 64 (lo32)
      g1[6] = 0;
      g1[7] = 0;
      v4i gz4 = {0, 0, 0, 0};
      v8i gz8 = {0, 0, 0, 0, 0, 0, 0, 0};
      __builtin_amdgcn_tensor_load_to_lds(g0, g1, gz4, gz4, gz8, 0);
      wait_tensorcnt0();
    }
#else
    for (int i = tid; i < BKV * (Dh / 2); i += 256) {
      int r = i >> 5, cp = i & 31;
      *(unsigned int*)&Ks[r][cp * 2] =
          ((const unsigned int*)Kh)[(size_t)(c0 + r) * (Dh / 2) + cp];
    }
#endif
    // ---- load V (bf16 pairs, coalesced) and transpose into [d][kv] ------
    for (int i = tid; i < BKV * (Dh / 2); i += 256) {
      int r = i >> 5, dp = i & 31;                 // kv row, d-pair index
      unsigned pr = ((const unsigned int*)Vh)[(size_t)(c0 + r) * (Dh / 2) + dp];
      __bf16 v0, v1;
      __builtin_memcpy(&v0, &pr, 2);
      char* prh = (char*)&pr + 2;
      __builtin_memcpy(&v1, prh, 2);
      Vt[2 * dp + 0][r] = v0;
      Vt[2 * dp + 1][r] = v1;
    }
    __syncthreads();

    // ---- S = Q @ K^T : 16(q rows) x 64(keys) per wave -------------------
    v8f sc[4];
#pragma unroll
    for (int nt = 0; nt < 4; ++nt) {
      sc[nt] = zero8();
#pragma unroll
      for (int ks = 0; ks < 2; ++ks) {
        v16bf kf;
        const int kr = nt * 16 + ln;          // key row = B column
        const int kh = ks * 32 + hi * 16;
#pragma unroll
        for (int i = 0; i < 16; ++i) kf[i] = Ks[kr][kh + i];
        sc[nt] = __builtin_amdgcn_wmma_f32_16x16x32_bf16(
            false, qa[ks], false, kf, (short)0, sc[nt], false, false);
      }
    }

    // ---- online softmax: VGPR slot r <-> Q row (r + 8*hi) ---------------
#pragma unroll
    for (int r = 0; r < 8; ++r) {
      float t = fmaxf(fmaxf(sc[0][r], sc[1][r]), fmaxf(sc[2][r], sc[3][r]));
      t = fmaxf(t, lane16_xor<8>(t));
      t = fmaxf(t, lane16_xor<4>(t));
      t = fmaxf(t, lane16_xor<2>(t));
      t = fmaxf(t, lane16_xor<1>(t));
      const float mn = fmaxf(mrow[r], t);
      const float al = fast_exp2(mrow[r] - mn);
      float rs = 0.f;
#pragma unroll
      for (int nt = 0; nt < 4; ++nt) {
        const float p = fast_exp2(sc[nt][r] - mn);
        sc[nt][r] = p;
        rs += p;
      }
      rs += lane16_xor<8>(rs);
      rs += lane16_xor<4>(rs);
      rs += lane16_xor<2>(rs);
      rs += lane16_xor<1>(rs);
      lrow[r] = lrow[r] * al + rs;
      mrow[r] = mn;
#pragma unroll
      for (int j = 0; j < 4; ++j) o[j][r] *= al;
    }

    // ---- bounce P through LDS (wave-private rows, no barrier needed) ----
#pragma unroll
    for (int nt = 0; nt < 4; ++nt) {
      const int col = nt * 16 + ln;
#pragma unroll
      for (int r = 0; r < 8; ++r)
        Ps[wv * 16 + r + hi * 8][col] = (__bf16)sc[nt][r];
    }

    // ---- O += P @ V -----------------------------------------------------
#pragma unroll
    for (int ks = 0; ks < 2; ++ks) {
      v16bf pa;
      const int row = wv * 16 + ln;
      const int kh = ks * 32 + hi * 16;
#pragma unroll
      for (int i = 0; i < 16; ++i) pa[i] = Ps[row][kh + i];
      const int kb2 = ks * 32 + hi * 16;
#pragma unroll
      for (int j = 0; j < 4; ++j) {
        v16bf vf;
        const int dcol = j * 16 + ln;
#pragma unroll
        for (int i = 0; i < 16; ++i) vf[i] = Vt[dcol][kb2 + i];
        o[j] = __builtin_amdgcn_wmma_f32_16x16x32_bf16(
            false, pa, false, vf, (short)0, o[j], false, false);
      }
    }
    __syncthreads();   // protect Ks/Vt for next chunk
  }

  // normalize and write in final (b, s, h*64+d) layout
  const int b = bh >> 4, h = bh & 15;
#pragma unroll
  for (int j = 0; j < 4; ++j) {
    const int dcol = j * 16 + ln;
#pragma unroll
    for (int r = 0; r < 8; ++r) {
      const int srow = q0 + wv * 16 + r + hi * 8;
      const float v = o[j][r] / lrow[r];
      Xh[(size_t)b * S * E + (size_t)srow * E + h * 64 + dcol] = (__bf16)v;
    }
  }
}

// ---------------------------------------------------------------------------
// Host-side launch
// ---------------------------------------------------------------------------
extern "C" void kernel_launch(void* const* d_in, const int* in_sizes, int n_in,
                              void* d_out, int out_size, void* d_ws, size_t ws_size,
                              hipStream_t stream) {
  (void)in_sizes; (void)n_in; (void)out_size; (void)ws_size;
  const float* x  = (const float*)d_in[0];
  const float* Wq = (const float*)d_in[1];
  const float* bq = (const float*)d_in[2];
  const float* Wk = (const float*)d_in[3];
  const float* bk = (const float*)d_in[4];
  const float* Wv = (const float*)d_in[5];
  const float* bv = (const float*)d_in[6];
  const float* Wo = (const float*)d_in[7];
  const float* bo = (const float*)d_in[8];
  float* out = (float*)d_out;

  const size_t slab = (size_t)BB * S * E;        // 4M bf16 elements per buffer
  __bf16* Qbuf = (__bf16*)d_ws;
  __bf16* Kbuf = Qbuf + slab;
  __bf16* Vbuf = Kbuf + slab;
  __bf16* Xh   = Vbuf + slab;                    // total 32 MB of workspace

  const int M = BB * S;                          // 4096
  dim3 blk(256);
  dim3 gGemm(E / 64, M / 128);                   // (16, 32)
  dim3 gAttn(S / 128, BB * H);                   // (16, 32)

  // Q/K/V projections (Q gets softmax scale * log2(e) folded in)
  hipLaunchKernelGGL((gemm_bias_wmma<false, true>), gGemm, blk, 0, stream,
                     (const void*)x, Wq, bq, (void*)Qbuf, M, E, E, QSCALE);
  hipLaunchKernelGGL((gemm_bias_wmma<false, true>), gGemm, blk, 0, stream,
                     (const void*)x, Wk, bk, (void*)Kbuf, M, E, E, 1.0f);
  hipLaunchKernelGGL((gemm_bias_wmma<false, true>), gGemm, blk, 0, stream,
                     (const void*)x, Wv, bv, (void*)Vbuf, M, E, E, 1.0f);

  // flash attention per (b,h)
  hipLaunchKernelGGL(flash_attn_wmma, gAttn, blk, 0, stream, Qbuf, Kbuf, Vbuf, Xh);

  // output projection -> fp32 d_out
  hipLaunchKernelGGL((gemm_bias_wmma<true, false>), gGemm, blk, 0, stream,
                     (const void*)Xh, Wo, bo, (void*)out, M, E, E, 1.0f);
}
